// SingleLayerDecoder_84636625535252
// MI455X (gfx1250) — matrix operand order repaired and checked by reading, not compiled, not run on hardware
//
#include <hip/hip_runtime.h>

typedef __bf16 v16bf __attribute__((ext_vector_type(16)));
typedef float  v8f   __attribute__((ext_vector_type(8)));

#define BSZ   256
#define TLEN  512
#define INS   1024
#define STATE 64
#define MTOT  (BSZ * TLEN)   // 131072 rows in the big GEMM

// ---------------------------------------------------------------------------
// Kernel 0: convert weights to bf16 (W_in, W_rec as-is; W_out zero-padded to
// a 16x64 block laid out column-of-B-major so B fragments are contiguous).
// ---------------------------------------------------------------------------
__global__ void cvt_weights_kernel(const float* __restrict__ W_in,
                                   const float* __restrict__ W_rec,
                                   const float* __restrict__ W_out,
                                   __bf16* __restrict__ wbin,
                                   __bf16* __restrict__ wbrec,
                                   __bf16* __restrict__ woutb) {
  int gid = blockIdx.x * blockDim.x + threadIdx.x;
  if (gid < STATE * INS)   wbin[gid]  = (__bf16)W_in[gid];
  if (gid < STATE * STATE) wbrec[gid] = (__bf16)W_rec[gid];
  if (gid < 16 * STATE) {
    int n = gid / STATE, k = gid % STATE;
    woutb[gid] = (__bf16)((n < 2) ? W_out[n * STATE + k] : 0.0f);
  }
}

// Pack 8 fp32 (two float4) into bf16 vector elements [base, base+8)
__device__ __forceinline__ void fill8(v16bf& a, int base, float4 u, float4 v) {
  a[base + 0] = (__bf16)u.x; a[base + 1] = (__bf16)u.y;
  a[base + 2] = (__bf16)u.z; a[base + 3] = (__bf16)u.w;
  a[base + 4] = (__bf16)v.x; a[base + 5] = (__bf16)v.y;
  a[base + 6] = (__bf16)v.z; a[base + 7] = (__bf16)v.w;
}

__device__ __forceinline__ void fill8sum(v16bf& a, int base,
                                         float4 e0, float4 s0,
                                         float4 e1, float4 s1) {
  a[base + 0] = (__bf16)(e0.x + s0.x); a[base + 1] = (__bf16)(e0.y + s0.y);
  a[base + 2] = (__bf16)(e0.z + s0.z); a[base + 3] = (__bf16)(e0.w + s0.w);
  a[base + 4] = (__bf16)(e1.x + s1.x); a[base + 5] = (__bf16)(e1.y + s1.y);
  a[base + 6] = (__bf16)(e1.z + s1.z); a[base + 7] = (__bf16)(e1.w + s1.w);
}

#define WMMA_BF16(A, B, C) \
  __builtin_amdgcn_wmma_f32_16x16x32_bf16(false, (A), false, (B), (short)0, (C), false, false)

// Raw (pre-conversion) A-fragment: 16 fp32 per lane in contiguous k order (k = 16*lh + s)
struct Afrag { float4 f[4]; };

__device__ __forceinline__ Afrag loadA(const float* __restrict__ row, int kc, int lh) {
  Afrag r;
  r.f[0] = *(const float4*)(row + kc + lh * 16);
  r.f[1] = *(const float4*)(row + kc + lh * 16 + 4);
  r.f[2] = *(const float4*)(row + kc + lh * 16 + 8);
  r.f[3] = *(const float4*)(row + kc + lh * 16 + 12);
  return r;
}

__device__ __forceinline__ v16bf cvtA(const Afrag& r) {
  v16bf a;
  fill8(a, 0, r.f[0], r.f[1]);
  fill8(a, 8, r.f[2], r.f[3]);
  return a;
}

__device__ __forceinline__ void loadB4(const __bf16* __restrict__ wbin, int kc,
                                       int lm, int lh, v16bf b[4]) {
  const __bf16* bp = wbin + kc + lh * 16;
#pragma unroll
  for (int nt = 0; nt < 4; ++nt)
    b[nt] = *(const v16bf*)(bp + (size_t)(nt * 16 + lm) * INS);
}

// ---------------------------------------------------------------------------
// Kernel 1: ext[M, 64] = x[M, 1024] @ W_in^T   (M = BS*T), bf16 WMMA, f32 acc.
// One wave per 16-row M-tile; software-pipelined K loop: loads for chunk kc+32
// are in flight while chunk kc is converted and fed to 4 WMMAs, so the inner
// loop never waits loadcnt to 0 on freshly issued loads. HBM-bound (~544 MB).
// ---------------------------------------------------------------------------
__global__ void __launch_bounds__(256)
proj_kernel(const float* __restrict__ x,
            const __bf16* __restrict__ wbin,
            float* __restrict__ ext) {
  const int lane = threadIdx.x & 31;
  const int lm = lane & 15;
  const int lh = lane >> 4;
  const int wave  = (blockIdx.x * blockDim.x + threadIdx.x) >> 5;
  const int mbase = wave * 16;

  v8f acc[4] = {};
  const float* xrow = x + (size_t)(mbase + lm) * INS;

  // Prologue: stage chunk 0
  Afrag a0 = loadA(xrow, 0, lh);
  v16bf b0[4];
  loadB4(wbin, 0, lm, lh, b0);

  for (int kc = 0; kc < INS - 32; kc += 32) {
    // Issue next chunk's loads first (overlap with this chunk's compute)
    Afrag a1 = loadA(xrow, kc + 32, lh);
    v16bf b1[4];
    loadB4(wbin, kc + 32, lm, lh, b1);

    v16bf a = cvtA(a0);
#pragma unroll
    for (int nt = 0; nt < 4; ++nt)
      acc[nt] = WMMA_BF16(a, b0[nt], acc[nt]);

    a0 = a1;
#pragma unroll
    for (int nt = 0; nt < 4; ++nt) b0[nt] = b1[nt];
  }
  // Epilogue: last chunk
  {
    v16bf a = cvtA(a0);
#pragma unroll
    for (int nt = 0; nt < 4; ++nt)
      acc[nt] = WMMA_BF16(a, b0[nt], acc[nt]);
  }

  // Store D: lane holds col nt*16+lm, rows v + 8*lh  (two 64B segments/row)
#pragma unroll
  for (int nt = 0; nt < 4; ++nt)
#pragma unroll
    for (int v = 0; v < 8; ++v)
      ext[(size_t)(mbase + v + 8 * lh) * STATE + nt * 16 + lm] = acc[nt][v];
}

// ---------------------------------------------------------------------------
// Kernel 2: serial recurrence + fused output projection.
// 16 blocks x 1 wave; each wave owns a 16-row batch tile.
// W_rec lives in registers as 8 pre-laid-out B fragments; state round-trips
// through a bank-conflict-free padded LDS tile (row stride 68 floats) to
// convert D-layout -> A-layout. The state tile is read from LDS exactly once
// per step: the same raw loads feed both (a) the W_out projection of the
// previous step and (b) the state+ext A fragment of the current step.
// ---------------------------------------------------------------------------
__global__ void __launch_bounds__(32)
recur_kernel(const float* __restrict__ ext,
             const __bf16* __restrict__ wbrec,
             const __bf16* __restrict__ woutb,
             float* __restrict__ out) {
  const int lane = threadIdx.x & 31;
  const int lm = lane & 15;
  const int lh = lane >> 4;
  const int b0 = blockIdx.x * 16;

  __shared__ float st[16 * 68];            // row stride 68 floats: conflict-free
  for (int i = threadIdx.x; i < 16 * 68; i += 32) st[i] = 0.0f;
  __syncthreads();

  // Preload W_rec^T B fragments: [nt][kc], col n = nt*16+lm -> W_rec row n
  v16bf brec[4][2];
#pragma unroll
  for (int nt = 0; nt < 4; ++nt)
#pragma unroll
    for (int kc = 0; kc < 2; ++kc)
      brec[nt][kc] =
          *(const v16bf*)(wbrec + (nt * 16 + lm) * STATE + kc * 32 + lh * 16);

  // Padded W_out^T B fragments (only cols 0,1 nonzero)
  v16bf bout[2];
#pragma unroll
  for (int kc = 0; kc < 2; ++kc)
    bout[kc] = *(const v16bf*)(woutb + lm * STATE + kc * 32 + lh * 16);

  const float* erow = ext + (size_t)(b0 + lm) * (TLEN * STATE);
  const float* srow = st + lm * 68;        // this lane's state row (A layout)

  for (int t = 0; t < TLEN; ++t) {
    // --- single LDS read pass: s_{t-1} in A layout -----------------------
    Afrag s[2];
#pragma unroll
    for (int kc = 0; kc < 2; ++kc)
      s[kc] = loadA(srow, kc * 32, lh);

    // --- output projection of step t-1 (off the recurrence critical path) -
    if (t > 0) {
      v16bf as0 = cvtA(s[0]);
      v16bf as1 = cvtA(s[1]);
      v8f occ = {};
      occ = WMMA_BF16(as0, bout[0], occ);
      occ = WMMA_BF16(as1, bout[1], occ);
      if (lm < 2) {   // D cols 0,1 hold the two outputs
#pragma unroll
        for (int v = 0; v < 8; ++v)
          out[((size_t)(b0 + v + 8 * lh) * TLEN + (t - 1)) * 2 + lm] = occ[v];
      }
    }

    // --- A = bf16(s_{t-1} + e_t) in contiguous k order --------------------------
    const float* ep = erow + t * STATE;
    v16bf a[2];
#pragma unroll
    for (int kc = 0; kc < 2; ++kc) {
      float4 e0 = *(const float4*)(ep + kc * 32 + lh * 16);
      float4 e1 = *(const float4*)(ep + kc * 32 + lh * 16 + 4);
      float4 e2 = *(const float4*)(ep + kc * 32 + lh * 16 + 8);
      float4 e3 = *(const float4*)(ep + kc * 32 + lh * 16 + 12);
      fill8sum(a[kc], 0, e0, s[kc].f[0], e1, s[kc].f[1]);
      fill8sum(a[kc], 8, e2, s[kc].f[2], e3, s[kc].f[3]);
    }

    // Hide L2 latency of the next timestep behind the WMMA chain
    if (t + 1 < TLEN) {
      __builtin_prefetch(erow + (t + 1) * STATE, 0, 0);
      __builtin_prefetch(erow + (t + 1) * STATE + 32, 0, 0);
    }

    // --- s_t = relu((s_{t-1} + e_t) @ W_rec^T) ----------------------------
    v8f acc[4];
#pragma unroll
    for (int nt = 0; nt < 4; ++nt) {
      v8f c = {};
      c = WMMA_BF16(a[0], brec[nt][0], c);
      c = WMMA_BF16(a[1], brec[nt][1], c);
      acc[nt] = c;
    }
#pragma unroll
    for (int nt = 0; nt < 4; ++nt)
#pragma unroll
      for (int v = 0; v < 8; ++v)
        acc[nt][v] = fmaxf(acc[nt][v], 0.0f);

    // --- single LDS write pass: D layout -> state tile --------------------
    __syncthreads();
#pragma unroll
    for (int nt = 0; nt < 4; ++nt)
#pragma unroll
      for (int v = 0; v < 8; ++v)
        st[(v + 8 * lh) * 68 + nt * 16 + lm] = acc[nt][v];
    __syncthreads();
  }

  // Epilogue: project the final state for out[:, T-1, :]
  {
    Afrag s[2];
#pragma unroll
    for (int kc = 0; kc < 2; ++kc)
      s[kc] = loadA(srow, kc * 32, lh);
    v16bf as0 = cvtA(s[0]);
    v16bf as1 = cvtA(s[1]);
    v8f occ = {};
    occ = WMMA_BF16(as0, bout[0], occ);
    occ = WMMA_BF16(as1, bout[1], occ);
    if (lm < 2) {
#pragma unroll
      for (int v = 0; v < 8; ++v)
        out[((size_t)(b0 + v + 8 * lh) * TLEN + (TLEN - 1)) * 2 + lm] = occ[v];
    }
  }
}

// ---------------------------------------------------------------------------
// Host launcher
// ---------------------------------------------------------------------------
extern "C" void kernel_launch(void* const* d_in, const int* in_sizes, int n_in,
                              void* d_out, int out_size, void* d_ws, size_t ws_size,
                              hipStream_t stream) {
  (void)in_sizes; (void)n_in; (void)out_size; (void)ws_size;
  const float* x     = (const float*)d_in[0];
  const float* W_in  = (const float*)d_in[1];
  const float* W_rec = (const float*)d_in[2];
  const float* W_out = (const float*)d_in[3];
  float* out = (float*)d_out;

  // Workspace layout (16B-aligned):
  //   [0, 32MB)        : ext   (fp32, [BS*T, 64])
  //   [+0, +128KB)     : wbin  (bf16 [64,1024])
  //   [+128KB, +8KB)   : wbrec (bf16 [64,64])
  //   [.., +2KB)       : woutb (bf16 [16,64], padded W_out^T columns)
  char* base = (char*)d_ws;
  float*  ext   = (float*)base;
  __bf16* wbin  = (__bf16*)(base + (size_t)MTOT * STATE * sizeof(float));
  __bf16* wbrec = (__bf16*)((char*)wbin  + (size_t)STATE * INS * sizeof(__bf16));
  __bf16* woutb = (__bf16*)((char*)wbrec + (size_t)STATE * STATE * sizeof(__bf16));

  // 0) weight conversion (covers 65536 elements; smaller arrays nested)
  cvt_weights_kernel<<<(STATE * INS + 255) / 256, 256, 0, stream>>>(
      W_in, W_rec, W_out, wbin, wbrec, woutb);

  // 1) big input projection: 8192 M-tiles, 8 waves/block
  proj_kernel<<<MTOT / 16 / 8, 256, 0, stream>>>(x, wbin, ext);

  // 2) serial recurrence + fused output projection: 16 batch tiles
  recur_kernel<<<BSZ / 16, 32, 0, stream>>>(ext, wbrec, woutb, out);
}
